// BasicSGNNClassifier_6571299963162
// MI455X (gfx1250) — matrix operand order repaired and checked by reading, not compiled, not run on hardware
//
#include <hip/hip_runtime.h>
#include <hip/hip_bf16.h>

typedef __attribute__((ext_vector_type(16))) _Float16 v16h;
typedef __attribute__((ext_vector_type(8)))  _Float16 v8h;
typedef __attribute__((ext_vector_type(2)))  _Float16 v2h;
typedef __attribute__((ext_vector_type(8)))  float    v8f;

#define NNODES 131072
#define NEDGES 2097152
#define FEAT   64
#define NGRAPH 256
#define NPGRP  512
#define TSTEP  8
#define RCHUNK 64          // NPGRP / TSTEP
#define NCLS   10
#define LSTRIDE 72         // LDS row stride in halves: 144 B, 16B-aligned, bank-spread

// ---------------------------------------------------------------- init
__global__ void k_init(float* __restrict__ agg, float* __restrict__ deg) {
  unsigned i = blockIdx.x * 256u + threadIdx.x;
  if (i < (unsigned)NNODES * FEAT) agg[i] = 0.0f;
  if (i < NNODES) deg[i] = 1.0f;     // self-loop contribution to degree
}

// ---------------------------------------------------------------- degree
__global__ void k_degree(const int* __restrict__ dst, float* __restrict__ deg) {
  unsigned e = blockIdx.x * 256u + threadIdx.x;
  if (e < NEDGES) atomicAdd(&deg[dst[e]], 1.0f);
}

__global__ void k_dinv(float* __restrict__ deg) {
  unsigned i = blockIdx.x * 256u + threadIdx.x;
  if (i < NNODES) deg[i] = rsqrtf(deg[i]);   // deg >= 1 always
}

// pack two input elements into two halves (f32 or f16 source)
__device__ __forceinline__ v2h pack2(const float* p) {
  float2 v = *(const float2*)p;
  v2h r; r[0] = (_Float16)v.x; r[1] = (_Float16)v.y; return r;
}
__device__ __forceinline__ v2h pack2(const _Float16* p) {
  return *(const v2h*)p;
}

// ---------------------------------------------------------------- WMMA GEMM: out[N,64] = in[N,64] @ W[64,64]
// One wave = 16-row tile. A (16x32 f16) fragments and transposed-W B fragments
// are laid out in LDS so each lane's fragment = contiguous 8-half runs
// -> ds_load_b128 only. 8 v_wmma_f32_16x16x32_f16 per tile.
template <typename TIn>
__global__ void __launch_bounds__(256) k_gemm64(const TIn* __restrict__ in,
                                                const float* __restrict__ W,
                                                float* __restrict__ out) {
  __shared__ __align__(16) _Float16 sWt[64 * LSTRIDE];      // sWt[col*LSTRIDE + k] = W[k][col]
  __shared__ __align__(16) _Float16 sA[8][16 * LSTRIDE];    // sA[w][row*LSTRIDE + k]
  const int tid  = threadIdx.x;
  const int wave = tid >> 5, lane = tid & 31;
  const int half = lane >> 4, l16 = lane & 15;
  const int rowBase = blockIdx.x * 128 + wave * 16;

  // stage W transposed: thread j handles col c = j>>5, k-pair kp = (j&31)*2
  for (int j = tid; j < 64 * 32; j += 256) {
    int c = j >> 5, kp = (j & 31) * 2;
    v2h p; p[0] = (_Float16)W[kp * 64 + c]; p[1] = (_Float16)W[(kp + 1) * 64 + c];
    *(v2h*)&sWt[c * LSTRIDE + kp] = p;
  }
  // stage A tile (16 rows x 64), packed 2-at-a-time
  for (int j = lane; j < 16 * 32; j += 32) {
    int r = j >> 5, cp = (j & 31) * 2;
    *(v2h*)&sA[wave][r * LSTRIDE + cp] = pack2(&in[(size_t)(rowBase + r) * FEAT + cp]);
  }
  __syncthreads();

  // A fragments: k(e) = 8*half + 16*(e>=8) + (e&7)  -> contiguous 8-half runs
  const _Float16* rowp = &sA[wave][l16 * LSTRIDE + half * 8];
  v8h a0lo = *(const v8h*)(rowp);
  v8h a0hi = *(const v8h*)(rowp + 16);
  v8h a1lo = *(const v8h*)(rowp + 32);
  v8h a1hi = *(const v8h*)(rowp + 48);
  v16h a0 = __builtin_shufflevector(a0lo, a0hi, 0,1,2,3,4,5,6,7,8,9,10,11,12,13,14,15);
  v16h a1 = __builtin_shufflevector(a1lo, a1hi, 0,1,2,3,4,5,6,7,8,9,10,11,12,13,14,15);

#pragma unroll
  for (int ct = 0; ct < 4; ++ct) {
    // B fragments: col = l16, k(e) = 8*half + 16*(e>=8) + (e&7)  -> contiguous 8-half runs in transposed W
    const _Float16* colp = &sWt[(ct * 16 + l16) * LSTRIDE + half * 8];
    v8h b0lo = *(const v8h*)(colp);
    v8h b0hi = *(const v8h*)(colp + 16);
    v8h b1lo = *(const v8h*)(colp + 32);
    v8h b1hi = *(const v8h*)(colp + 48);
    v16h b0 = __builtin_shufflevector(b0lo, b0hi, 0,1,2,3,4,5,6,7,8,9,10,11,12,13,14,15);
    v16h b1 = __builtin_shufflevector(b1lo, b1hi, 0,1,2,3,4,5,6,7,8,9,10,11,12,13,14,15);

    v8f c = {0.f, 0.f, 0.f, 0.f, 0.f, 0.f, 0.f, 0.f};
    c = __builtin_amdgcn_wmma_f32_16x16x32_f16(false, a0, false, b0, (short)0, c, false, false);
    c = __builtin_amdgcn_wmma_f32_16x16x32_f16(false, a1, false, b1, (short)0, c, false, false);
    // D: row = 8*half + v, col = l16
#pragma unroll
    for (int v = 0; v < 8; ++v)
      out[(size_t)(rowBase + half * 8 + v) * FEAT + ct * 16 + l16] = c[v];
  }
}

// ---------------------------------------------------------------- edge scatter (32 lanes per edge, L2 atomics)
__global__ void k_scatter(const int* __restrict__ src, const int* __restrict__ dst,
                          const float* __restrict__ dinv, const float* __restrict__ h,
                          float* __restrict__ agg) {
  unsigned gid = blockIdx.x * 256u + threadIdx.x;
  unsigned e = gid >> 5, lane = gid & 31;
  if (e >= NEDGES) return;
  int s = src[e], d = dst[e];
  float coef = dinv[s] * dinv[d];
  float v0 = h[(size_t)s * FEAT + lane] * coef;
  float v1 = h[(size_t)s * FEAT + 32 + lane] * coef;
  atomicAdd(&agg[(size_t)d * FEAT + lane], v0);
  atomicAdd(&agg[(size_t)d * FEAT + 32 + lane], v1);
}

// ---------------------------------------------------------------- conv1 epilogue: self-loop + bias + relu -> f16; re-zero agg
__global__ void k_epilogue1(float* __restrict__ agg, const float* __restrict__ h,
                            const float* __restrict__ dinv, const float* __restrict__ b,
                            _Float16* __restrict__ r16) {
  unsigned i = blockIdx.x * 256u + threadIdx.x;
  if (i >= (unsigned)NNODES * FEAT) return;
  unsigned node = i >> 6, f = i & 63;
  float dv = dinv[node];
  float v = agg[i] + h[i] * dv * dv + b[f];
  r16[i] = (_Float16)(v > 0.f ? v : 0.f);
  agg[i] = 0.0f;                      // ready for conv2 scatter
}

// ---------------------------------------------------------------- node_blur fused with conv2 epilogue
__global__ void k_blur(const float* __restrict__ agg, const float* __restrict__ h,
                       const float* __restrict__ dinv, const float* __restrict__ b,
                       float* __restrict__ z) {
  int g = blockIdx.x;
  int t = threadIdx.x >> 6;
  int f = threadIdx.x & 63;
  float acc = 0.f;
  for (int r = 0; r < RCHUNK; ++r) {
    int node = g * NPGRP + r * TSTEP + t;
    float dv = dinv[node];
    size_t idx = (size_t)node * FEAT + f;
    float v = agg[idx] + h[idx] * dv * dv + b[f];
    acc += ((float)(RCHUNK - 1 - r) / (float)(RCHUNK - 1)) * v;   // linspace(1,0,64)
  }
  z[g * NPGRP + t * FEAT + f] = acc;   // matches reshape(G, T*HID)
}

// ---------------------------------------------------------------- LIF rate (input constant across the 4 steps)
__device__ __forceinline__ float lif_rate(float x) {
  float mem = 0.f, s = 0.f;
#pragma unroll
  for (int i = 0; i < 4; ++i) {
    float reset = (mem > 1.0f) ? 1.0f : 0.0f;
    mem = 0.9f * mem + x - reset;
    s += (mem > 1.0f) ? 1.0f : 0.0f;
  }
  return s * 0.25f;
}

// ---------------------------------------------------------------- classifier: one block per graph
__global__ void k_classify(const float* __restrict__ z,
                           const float* __restrict__ w1, const float* __restrict__ b1,
                           const float* __restrict__ w2, const float* __restrict__ b2,
                           const float* __restrict__ w3, const float* __restrict__ b3,
                           float* __restrict__ out) {
  __shared__ float sz[NPGRP];
  __shared__ float s1[64];
  __shared__ float s2[64];
  int g = blockIdx.x, tid = threadIdx.x;
  for (int j = tid; j < NPGRP; j += 64) sz[j] = z[g * NPGRP + j];
  __syncthreads();
  float acc = b1[tid];
  for (int k = 0; k < NPGRP; ++k) acc += sz[k] * w1[k * 64 + tid];
  s1[tid] = lif_rate(acc);
  __syncthreads();
  acc = b2[tid];
  for (int k = 0; k < 64; ++k) acc += s1[k] * w2[k * 64 + tid];
  s2[tid] = lif_rate(acc);
  __syncthreads();
  if (tid < NCLS) {
    acc = b3[tid];
    for (int k = 0; k < 64; ++k) acc += s2[k] * w3[k * NCLS + tid];
    out[g * NCLS + tid] = acc;
  }
}

// ---------------------------------------------------------------- launch
extern "C" void kernel_launch(void* const* d_in, const int* in_sizes, int n_in,
                              void* d_out, int out_size, void* d_ws, size_t ws_size,
                              hipStream_t stream) {
  const float* x       = (const float*)d_in[0];
  const int*   eidx    = (const int*)d_in[1];
  const int*   src     = eidx;
  const int*   dst     = eidx + NEDGES;
  const float* conv1_w = (const float*)d_in[3];
  const float* conv1_b = (const float*)d_in[4];
  const float* conv2_w = (const float*)d_in[5];
  const float* conv2_b = (const float*)d_in[6];
  const float* lin1_w  = (const float*)d_in[7];
  const float* lin1_b  = (const float*)d_in[8];
  const float* lin2_w  = (const float*)d_in[9];
  const float* lin2_b  = (const float*)d_in[10];
  const float* lin3_w  = (const float*)d_in[11];
  const float* lin3_b  = (const float*)d_in[12];

  char* ws = (char*)d_ws;
  size_t off = 0;
  auto alloc = [&](size_t bytes) {
    void* p = ws + off;
    off = (off + bytes + 255) & ~(size_t)255;
    return p;
  };
  float*    deg = (float*)   alloc((size_t)NNODES * 4);            // deg -> dinv in place
  float*    h   = (float*)   alloc((size_t)NNODES * FEAT * 4);     // GEMM outputs (reused)
  float*    agg = (float*)   alloc((size_t)NNODES * FEAT * 4);     // scatter accumulator (reused)
  _Float16* r16 = (_Float16*)alloc((size_t)NNODES * FEAT * 2);     // relu(conv1) in f16
  float*    z   = (float*)   alloc((size_t)NGRAPH * NPGRP * 4);    // blur output

  k_init    <<<(NNODES * FEAT) / 256, 256, 0, stream>>>(agg, deg);
  k_degree  <<<NEDGES / 256,          256, 0, stream>>>(dst, deg);
  k_dinv    <<<NNODES / 256,          256, 0, stream>>>(deg);
  k_gemm64<float>   <<<NNODES / 128,  256, 0, stream>>>(x, conv1_w, h);
  k_scatter <<<NEDGES / 8,            256, 0, stream>>>(src, dst, deg, h, agg);
  k_epilogue1<<<(NNODES * FEAT) / 256, 256, 0, stream>>>(agg, h, deg, conv1_b, r16);
  k_gemm64<_Float16><<<NNODES / 128,  256, 0, stream>>>(r16, conv2_w, h);
  k_scatter <<<NEDGES / 8,            256, 0, stream>>>(src, dst, deg, h, agg);
  k_blur    <<<NGRAPH,                512, 0, stream>>>(agg, h, deg, conv2_b, z);
  k_classify<<<NGRAPH, 64, 0, stream>>>(z, lin1_w, lin1_b, lin2_w, lin2_b,
                                        lin3_w, lin3_b, (float*)d_out);
}
